// AugmentedModel_70205535420721
// MI455X (gfx1250) — matrix operand order repaired and checked by reading, not compiled, not run on hardware
//
#include <hip/hip_runtime.h>
#include <hip/hip_bf16.h>
#include <math.h>

typedef __bf16 bf16;
typedef __attribute__((ext_vector_type(16))) __bf16 v16bf;
typedef __attribute__((ext_vector_type(8)))  float  v8f;

static constexpr int kNL  = 6;
static constexpr int kH   = 12;
static constexpr int kD   = 384;
static constexpr int kDK  = 32;
static constexpr int kDFF = 1536;
static constexpr int kL   = 512;
static constexpr int kB   = 64;
static constexpr int kM   = kB * kL;   // 32768 rows
static constexpr int kTS  = 40;        // padded LDS tile row stride (halfs): 80B, 16B-aligned, conflict-free

__device__ __forceinline__ float sigm_(float x) { return 1.0f / (1.0f + expf(-x)); }
__device__ __forceinline__ float gelu_(float x) {
  float x3 = x * x * x;
  return 0.5f * x * (1.0f + tanhf(0.7978845608028654f * (x + 0.044715f * x3)));
}

// ---- CDNA5 async global->LDS copy (ASYNCcnt path) ----
__device__ __forceinline__ void async_ld_b128(uint32_t lds_off, const void* gptr) {
  asm volatile("global_load_async_to_lds_b128 %0, %1, off"
               :: "v"(lds_off), "v"((unsigned long long)(uintptr_t)gptr)
               : "memory");
}
__device__ __forceinline__ void wait_async0() {
  asm volatile("s_wait_asynccnt 0x0" ::: "memory");
}

// ---------------- block reduction helper (128 threads) ----------------
__device__ __forceinline__ float blk_reduce128(float v, float* red, int tid) {
  red[tid] = v; __syncthreads();
  #pragma unroll
  for (int s = 64; s > 0; s >>= 1) { if (tid < s) red[tid] += red[tid + s]; __syncthreads(); }
  float r = red[0]; __syncthreads();
  return r;
}

// ---------------- embedding + LayerNorm ----------------
__global__ __launch_bounds__(128) void embed_ln_kernel(
    const int* __restrict__ ids, const float* __restrict__ wemb,
    const float* __restrict__ pemb, const float* __restrict__ temb,
    const float* __restrict__ g, const float* __restrict__ bb,
    float* __restrict__ X, bf16* __restrict__ XBF)
{
  __shared__ float buf[kD];
  __shared__ float red[128];
  const int row = blockIdx.x, tid = threadIdx.x;
  const int t = row % kL;
  const int tok = ids[row];
  float s = 0.f;
  for (int j = tid; j < kD; j += 128) {
    float v = wemb[(size_t)tok * kD + j] + pemb[(size_t)t * kD + j] + temb[j];
    buf[j] = v; s += v;
  }
  float mean = blk_reduce128(s, red, tid) * (1.0f / kD);
  float s2 = 0.f;
  for (int j = tid; j < kD; j += 128) { float d = buf[j] - mean; s2 += d * d; }
  float var = blk_reduce128(s2, red, tid) * (1.0f / kD);
  float rstd = rsqrtf(var + 1e-12f);
  for (int j = tid; j < kD; j += 128) {
    float o = (buf[j] - mean) * rstd * g[j] + bb[j];
    X[(size_t)row * kD + j] = o;
    XBF[(size_t)row * kD + j] = (bf16)o;
  }
}

// ---------------- residual add + LayerNorm (fp32 + bf16 out) ----------------
__global__ __launch_bounds__(128) void add_ln_kernel(
    const float* __restrict__ Xin, const float* __restrict__ Y,
    const float* __restrict__ g, const float* __restrict__ bb,
    float* __restrict__ Xout, bf16* __restrict__ XBF)
{
  __shared__ float buf[kD];
  __shared__ float red[128];
  const int row = blockIdx.x, tid = threadIdx.x;
  float s = 0.f;
  for (int j = tid; j < kD; j += 128) {
    float v = Xin[(size_t)row * kD + j] + Y[(size_t)row * kD + j];
    buf[j] = v; s += v;
  }
  float mean = blk_reduce128(s, red, tid) * (1.0f / kD);
  float s2 = 0.f;
  for (int j = tid; j < kD; j += 128) { float d = buf[j] - mean; s2 += d * d; }
  float var = blk_reduce128(s2, red, tid) * (1.0f / kD);
  float rstd = rsqrtf(var + 1e-12f);
  for (int j = tid; j < kD; j += 128) {
    float o = (buf[j] - mean) * rstd * g[j] + bb[j];
    Xout[(size_t)row * kD + j] = o;
    XBF[(size_t)row * kD + j] = (bf16)o;
  }
}

// ---------------- weight transpose + f32->bf16 : W[K][N] -> WT[N][K] ----------------
__global__ __launch_bounds__(256) void transpose_cvt_kernel(
    const float* __restrict__ W, bf16* __restrict__ WT, int K, int N)
{
  __shared__ float tile[32][33];
  const int n0 = blockIdx.x * 32, k0 = blockIdx.y * 32;
  for (int i = threadIdx.y; i < 32; i += 8)
    tile[i][threadIdx.x] = W[(size_t)(k0 + i) * N + n0 + threadIdx.x];
  __syncthreads();
  for (int i = threadIdx.y; i < 32; i += 8)
    WT[(size_t)(n0 + i) * K + k0 + threadIdx.x] = (bf16)tile[threadIdx.x][i];
}

// ---------------- WMMA bf16 GEMM: C[M,N] = A[M,K] @ BT[N,K]^T + bias ----------------
// 128x128x32 tile, 8 waves (2x4), 8 WMMA / wave / K-step, double-buffered LDS,
// tiles staged with global_load_async_to_lds_b128 (ASYNCcnt).
// MODE 0: f32 out (+bias). MODE 1: bf16 out, gelu(gelu(acc+bias)).
template <int MODE>
__global__ __launch_bounds__(256) void gemm_bf16_kernel(
    const bf16* __restrict__ A, const bf16* __restrict__ BT,
    const float* __restrict__ bias, void* __restrict__ Cout,
    int Kdim, int Ndim)
{
  __shared__ __align__(16) bf16 sA[2][128 * kTS];
  __shared__ __align__(16) bf16 sB[2][128 * kTS];
  const int tid = threadIdx.x;
  const int wave = tid >> 5, lane = tid & 31;
  const int lo = lane & 15, hi = lane >> 4;
  const int wm = wave >> 2, wn = wave & 3;          // 2 x 4 wave grid
  const int m0 = blockIdx.y * 128, n0 = blockIdx.x * 128;

  const uint32_t aBase = (uint32_t)(uintptr_t)(&sA[0][0]);
  const uint32_t bBase = (uint32_t)(uintptr_t)(&sB[0][0]);
  const uint32_t bufStride = 128 * kTS * 2;   // bytes per buffer

  v8f c[4][2];
  const v8f zf = {0.f, 0.f, 0.f, 0.f, 0.f, 0.f, 0.f, 0.f};
  #pragma unroll
  for (int i = 0; i < 4; i++)
    #pragma unroll
    for (int j = 0; j < 2; j++) c[i][j] = zf;

  const int nk = Kdim >> 5;

  auto issue_tile = [&](int buf, int k0) {
    #pragma unroll
    for (int i = 0; i < 2; i++) {               // 512 16B chunks per tile / 256 thr
      int idx = tid + i * 256;
      int r = idx >> 2, s = idx & 3;
      uint32_t loff = (uint32_t)(r * kTS + s * 8) * 2 + (uint32_t)buf * bufStride;
      async_ld_b128(aBase + loff, &A[(size_t)(m0 + r) * Kdim + k0 + s * 8]);
      async_ld_b128(bBase + loff, &BT[(size_t)(n0 + r) * Kdim + k0 + s * 8]);
    }
  };

  issue_tile(0, 0);
  wait_async0();
  __syncthreads();

  for (int kb = 0; kb < nk; kb++) {
    const int cur = kb & 1;
    if (kb + 1 < nk) issue_tile(cur ^ 1, (kb + 1) << 5);

    const bf16* la = &sA[cur][0];
    const bf16* lb = &sB[cur][0];
    union FragU { v16bf v; uint4 u[2]; } a[4], bfr[2];
    #pragma unroll
    for (int i = 0; i < 4; i++) {               // A frag: K split {0-7,16-23}/{8-15,24-31}
      int r = (wm * 4 + i) * 16 + lo;
      a[i].u[0] = *(const uint4*)&la[r * kTS + hi * 8];
      a[i].u[1] = *(const uint4*)&la[r * kTS + 16 + hi * 8];
    }
    #pragma unroll
    for (int j = 0; j < 2; j++) {               // B frag: K split {0-7,16-23}/{8-15,24-31}
      int n = (wn * 2 + j) * 16 + lo;
      bfr[j].u[0] = *(const uint4*)&lb[n * kTS + hi * 8];
      bfr[j].u[1] = *(const uint4*)&lb[n * kTS + 16 + hi * 8];
    }
    #pragma unroll
    for (int i = 0; i < 4; i++)
      #pragma unroll
      for (int j = 0; j < 2; j++)
        c[i][j] = __builtin_amdgcn_wmma_f32_16x16x32_bf16(
            false, a[i].v, false, bfr[j].v, (short)0, c[i][j], false, false);

    wait_async0();
    __syncthreads();
  }

  #pragma unroll
  for (int i = 0; i < 4; i++)
    #pragma unroll
    for (int j = 0; j < 2; j++) {
      int col = n0 + (wn * 2 + j) * 16 + lo;
      float bv = bias[col];
      #pragma unroll
      for (int r = 0; r < 8; r++) {
        int row = m0 + (wm * 4 + i) * 16 + hi * 8 + r;
        float val = c[i][j][r] + bv;
        if (MODE == 0) {
          ((float*)Cout)[(size_t)row * Ndim + col] = val;
        } else {
          val = gelu_(gelu_(val));
          ((bf16*)Cout)[(size_t)row * Ndim + col] = (bf16)val;
        }
      }
    }
}

// ---------------- per-head l2norm / mask + transpose [B,L,H,DK]->[B,H,L,DK] ----------
// mode 0: copy (v); mode 1: l2norm (q); mode 2: l2norm * mask (k)
__global__ __launch_bounds__(256) void qkprep_kernel(
    const float* __restrict__ src, float* __restrict__ dst,
    const float* __restrict__ mask, int mode)
{
  const int lane = threadIdx.x & 31;
  const int idx = blockIdx.x * 8 + (threadIdx.x >> 5);   // (b,l,h) flat
  const int h = idx % kH;
  const int tmp = idx / kH;
  const int l = tmp % kL;
  const int b = tmp / kL;
  float v = src[(size_t)(b * kL + l) * kD + h * kDK + lane];
  if (mode != 0) {
    float ss = v * v;
    #pragma unroll
    for (int o = 16; o > 0; o >>= 1) ss += __shfl_xor(ss, o, 32);
    v = v / (sqrtf(ss) + 1e-6f);
    if (mode == 2) v *= mask[b * kL + l];
  }
  dst[(((size_t)b * kH + h) * kL + l) * kDK + lane] = v;
}

// ---------------- beta = sigmoid(x @ Wb) * mask ----------------
__global__ __launch_bounds__(256) void beta_kernel(
    const float* __restrict__ X, const float* __restrict__ Wb,
    const float* __restrict__ mask, float* __restrict__ BETA)
{
  const int lane = threadIdx.x & 31;
  const int row = blockIdx.x * 8 + (threadIdx.x >> 5);
  float acc[kH];
  #pragma unroll
  for (int h = 0; h < kH; h++) acc[h] = 0.f;
  for (int j = lane; j < kD; j += 32) {
    float xv = X[(size_t)row * kD + j];
    const float* w = &Wb[(size_t)j * kH];
    #pragma unroll
    for (int h = 0; h < kH; h++) acc[h] += xv * w[h];
  }
  float out = 0.f;
  #pragma unroll
  for (int h = 0; h < kH; h++) {
    #pragma unroll
    for (int o = 16; o > 0; o >>= 1) acc[h] += __shfl_xor(acc[h], o, 32);
    out = (lane == h) ? acc[h] : out;
  }
  if (lane < kH) BETA[(size_t)row * kH + lane] = mask[row] * sigm_(out);
}

// ---------------- hierarchical delta-rule scan: one wave32 block per (b,h) ----------
// States register-resident; 4-way split accumulators; pipelined loads; bf16 out fused.
__global__ __launch_bounds__(32) void scan_kernel(
    const float* __restrict__ QT, const float* __restrict__ KT,
    const float* __restrict__ VT, const float* __restrict__ BETA,
    const float* __restrict__ decayF, const float* __restrict__ decayS,
    bf16* __restrict__ OBF)
{
  const int lane = threadIdx.x;                 // v index (state column)
  const int bh = blockIdx.x;
  const int b = bh / kH, h = bh % kH;
  const float gf = sigm_(decayF[h]);
  const float gs = sigm_(decayS[h]);
  const float* q = QT + (size_t)bh * kL * kDK;
  const float* k = KT + (size_t)bh * kL * kDK;
  const float* v = VT + (size_t)bh * kL * kDK;

  float Sf[kDK], Ss[kDK];
  #pragma unroll
  for (int j = 0; j < kDK; j++) { Sf[j] = 0.f; Ss[j] = 0.f; }

  float qt = q[lane], kt = k[lane], vt = v[lane];
  float bt = BETA[(size_t)(b * kL) * kH + h];

  for (int t = 0; t < kL; t++) {
    float qn = 0.f, kn = 0.f, vn = 0.f, bn = 0.f;
    if (t + 1 < kL) {                           // pipeline next step's operands
      qn = q[(t + 1) * kDK + lane];
      kn = k[(t + 1) * kDK + lane];
      vn = v[(t + 1) * kDK + lane];
      bn = BETA[(size_t)(b * kL + t + 1) * kH + h];
    }
    float kl[kDK];
    #pragma unroll
    for (int j = 0; j < kDK; j++) kl[j] = __shfl(kt, j, 32);

    float pf[4] = {0.f, 0.f, 0.f, 0.f}, ps[4] = {0.f, 0.f, 0.f, 0.f};
    #pragma unroll
    for (int j = 0; j < kDK; j++) { pf[j & 3] += kl[j] * Sf[j]; ps[j & 3] += kl[j] * Ss[j]; }
    const float pfs = (pf[0] + pf[1]) + (pf[2] + pf[3]);
    const float pss = (ps[0] + ps[1]) + (ps[2] + ps[3]);
    const float uf = bt * (vt - pfs);
    const float us = bt * (vt - pss);

    float of[4] = {0.f, 0.f, 0.f, 0.f}, os[4] = {0.f, 0.f, 0.f, 0.f};
    #pragma unroll
    for (int j = 0; j < kDK; j++) {
      float ql = __shfl(qt, j, 32);
      Sf[j] = gf * Sf[j] + kl[j] * uf;
      of[j & 3] += ql * Sf[j];
      Ss[j] = gs * Ss[j] + kl[j] * us;
      os[j & 3] += ql * Ss[j];
    }
    const float ov = 0.5f * (((of[0] + of[1]) + (of[2] + of[3])) +
                             ((os[0] + os[1]) + (os[2] + os[3])));
    OBF[(size_t)(b * kL + t) * kD + h * kDK + lane] = (bf16)ov;
    qt = qn; kt = kn; vt = vn; bt = bn;
  }
}

// ---------------- masked mean pool + l2 normalize ----------------
__global__ __launch_bounds__(384) void pool_norm_kernel(
    const float* __restrict__ X, const float* __restrict__ mask,
    float* __restrict__ out)
{
  __shared__ float red[kD];
  const int b = blockIdx.x, d = threadIdx.x;
  float s = 0.f, ms = 0.f;
  for (int l = 0; l < kL; l++) {
    float m = mask[b * kL + l];
    s += X[(size_t)(b * kL + l) * kD + d] * m;
    ms += m;
  }
  float emb = s / fmaxf(ms, 1e-9f);
  red[d] = emb * emb; __syncthreads();
  if (d < 128) red[d] += red[d + 128] + red[d + 256];
  __syncthreads();
  #pragma unroll
  for (int sft = 64; sft > 0; sft >>= 1) { if (d < sft) red[d] += red[d + sft]; __syncthreads(); }
  float n = fmaxf(sqrtf(red[0]), 1e-12f);
  out[(size_t)b * kD + d] = emb / n;
}

// =========================== launcher ===========================
extern "C" void kernel_launch(void* const* d_in, const int* in_sizes, int n_in,
                              void* d_out, int out_size, void* d_ws, size_t ws_size,
                              hipStream_t stream)
{
  (void)in_sizes; (void)n_in; (void)out_size; (void)ws_size;
  const int*   ids  = (const int*)  d_in[0];
  const float* mask = (const float*)d_in[1];
  const float* wemb = (const float*)d_in[2];
  const float* pemb = (const float*)d_in[3];
  const float* temb = (const float*)d_in[4];
  const float* eg   = (const float*)d_in[5];
  const float* ebv  = (const float*)d_in[6];
  const float* Wq   = (const float*)d_in[7];
  const float* bq   = (const float*)d_in[8];
  const float* Wk   = (const float*)d_in[9];
  const float* bk   = (const float*)d_in[10];
  const float* Wv   = (const float*)d_in[11];
  const float* bv   = (const float*)d_in[12];
  const float* Wb   = (const float*)d_in[13];
  const float* dF   = (const float*)d_in[14];
  const float* dS   = (const float*)d_in[15];
  const float* Wo   = (const float*)d_in[16];
  const float* bo   = (const float*)d_in[17];
  const float* l1g  = (const float*)d_in[18];
  const float* l1b  = (const float*)d_in[19];
  const float* W1   = (const float*)d_in[20];
  const float* b1   = (const float*)d_in[21];
  const float* W2   = (const float*)d_in[22];
  const float* b2   = (const float*)d_in[23];
  const float* l2g  = (const float*)d_in[24];
  const float* l2b  = (const float*)d_in[25];

  char* base = (char*)d_ws;
  size_t off = 0;
  auto carve = [&](size_t bytes) -> char* {
    char* p = base + off;
    off = (off + bytes + 255) & ~(size_t)255;
    return p;
  };
  float* X    = (float*)carve((size_t)kM * kD * 4);
  bf16*  XBF  = (bf16*) carve((size_t)kM * kD * 2);
  float* TMP  = (float*)carve((size_t)kM * kD * 4);
  float* QT   = (float*)carve((size_t)kM * kD * 4);
  float* KT   = (float*)carve((size_t)kM * kD * 4);
  float* VT   = (float*)carve((size_t)kM * kD * 4);
  float* BETA = (float*)carve((size_t)kM * kH * 4);
  bf16*  OBF  = (bf16*) carve((size_t)kM * kD * 2);
  bf16*  HBF  = (bf16*) carve((size_t)kM * kDFF * 2);
  bf16*  WqT  = (bf16*) carve((size_t)kD * kD * 2);
  bf16*  WkT  = (bf16*) carve((size_t)kD * kD * 2);
  bf16*  WvT  = (bf16*) carve((size_t)kD * kD * 2);
  bf16*  WoT  = (bf16*) carve((size_t)kD * kD * 2);
  bf16*  W1T  = (bf16*) carve((size_t)kD * kDFF * 2);
  bf16*  W2T  = (bf16*) carve((size_t)kDFF * kD * 2);

  embed_ln_kernel<<<kM, 128, 0, stream>>>(ids, wemb, pemb, temb, eg, ebv, X, XBF);

  const dim3 tb(32, 8);
  const dim3 gD(kD / 128, kM / 128);     // (3, 256)
  const dim3 gF(kDFF / 128, kM / 128);   // (12, 256)
  const int prepBlocks = (kB * kL * kH) / 8;

  for (int l = 0; l < kNL; l++) {
    transpose_cvt_kernel<<<dim3(kD / 32, kD / 32), tb, 0, stream>>>(Wq + (size_t)l * kD * kD, WqT, kD, kD);
    transpose_cvt_kernel<<<dim3(kD / 32, kD / 32), tb, 0, stream>>>(Wk + (size_t)l * kD * kD, WkT, kD, kD);
    transpose_cvt_kernel<<<dim3(kD / 32, kD / 32), tb, 0, stream>>>(Wv + (size_t)l * kD * kD, WvT, kD, kD);
    transpose_cvt_kernel<<<dim3(kD / 32, kD / 32), tb, 0, stream>>>(Wo + (size_t)l * kD * kD, WoT, kD, kD);
    transpose_cvt_kernel<<<dim3(kDFF / 32, kD / 32), tb, 0, stream>>>(W1 + (size_t)l * kD * kDFF, W1T, kD, kDFF);
    transpose_cvt_kernel<<<dim3(kD / 32, kDFF / 32), tb, 0, stream>>>(W2 + (size_t)l * kDFF * kD, W2T, kDFF, kD);

    gemm_bf16_kernel<0><<<gD, 256, 0, stream>>>(XBF, WqT, bq + l * kD, TMP, kD, kD);
    qkprep_kernel<<<prepBlocks, 256, 0, stream>>>(TMP, QT, mask, 1);
    gemm_bf16_kernel<0><<<gD, 256, 0, stream>>>(XBF, WkT, bk + l * kD, TMP, kD, kD);
    qkprep_kernel<<<prepBlocks, 256, 0, stream>>>(TMP, KT, mask, 2);
    gemm_bf16_kernel<0><<<gD, 256, 0, stream>>>(XBF, WvT, bv + l * kD, TMP, kD, kD);
    qkprep_kernel<<<prepBlocks, 256, 0, stream>>>(TMP, VT, mask, 0);

    beta_kernel<<<kM / 8, 256, 0, stream>>>(X, Wb + (size_t)l * kD * kH, mask, BETA);
    scan_kernel<<<kB * kH, 32, 0, stream>>>(QT, KT, VT, BETA, dF + l * kH, dS + l * kH, OBF);

    gemm_bf16_kernel<0><<<gD, 256, 0, stream>>>(OBF, WoT, bo + l * kD, TMP, kD, kD);
    add_ln_kernel<<<kM, 128, 0, stream>>>(X, TMP, l1g + l * kD, l1b + l * kD, X, XBF);

    gemm_bf16_kernel<1><<<gF, 256, 0, stream>>>(XBF, W1T, b1 + l * kDFF, HBF, kD, kDFF);
    gemm_bf16_kernel<0><<<gD, 256, 0, stream>>>(HBF, W2T, b2 + l * kD, TMP, kDFF, kD);
    add_ln_kernel<<<kM, 128, 0, stream>>>(X, TMP, l2g + l * kD, l2b + l * kD, X, XBF);
  }

  pool_norm_kernel<<<kB, kD, 0, stream>>>(X, mask, (float*)d_out);
}
